// ATDTransformerLayer_24532853194954
// MI455X (gfx1250) — compile-verified, matrix operand order checked by reading, not run on hardware
//
#include <hip/hip_runtime.h>
#include <math.h>
#include <stdint.h>

typedef __attribute__((ext_vector_type(16))) _Float16 v16h;
typedef __attribute__((ext_vector_type(8)))  _Float16 v8h;
typedef __attribute__((ext_vector_type(8)))  float    v8f;
typedef __attribute__((ext_vector_type(4)))  int      v4i;
typedef __attribute__((ext_vector_type(8)))  int      v8i;

#define B_     2
#define H_     128
#define W_     128
#define N_     (H_*W_)
#define BN_    (B_*N_)
#define DIM_   192
#define C3_    (3*DIM_)
#define HEADS_ 6
#define HD_    32
#define WS_    16
#define SHIFT_ 8
#define NWIN_  64
#define NW2_   256
#define NT_    128
#define RD_    20
#define HID_   384

__device__ __forceinline__ float gelu_exact(float x) {
    return 0.5f * x * (1.0f + erff(x * 0.70710678118654752f));
}

// load a 16x32 f16 A/B-fragment row slice from LDS: e0..7 = [kb..kb+7], e8..15 = [kb+16..kb+23]
__device__ __forceinline__ v16h load_frag_lds(const _Float16* row, int kb) {
    v8h lo = *(const v8h*)(row + kb);
    v8h hi = *(const v8h*)(row + kb + 16);
    v16h f;
#pragma unroll
    for (int e = 0; e < 8; ++e) { f[e] = lo[e]; f[8 + e] = hi[e]; }
    return f;
}

// ---------------------------------------------------------------- convert f32 -> f16
__global__ void cvt_f32_f16_kernel(const float* __restrict__ src, _Float16* __restrict__ dst, long n) {
    long i = (long)blockIdx.x * blockDim.x + threadIdx.x;
    if (i < n) dst[i] = (_Float16)src[i];
}

// ---------------------------------------------------------------- LayerNorm over DIM_ (192), write f16
__global__ __launch_bounds__(256)
void ln_kernel(const float* __restrict__ x, const float* __restrict__ w,
               const float* __restrict__ b, _Float16* __restrict__ outh) {
    __shared__ float red[256];
    long row = blockIdx.x;
    int tid = threadIdx.x;
    float v = (tid < DIM_) ? x[row * DIM_ + tid] : 0.0f;
    red[tid] = v; __syncthreads();
    for (int off = 128; off >= 1; off >>= 1) { if (tid < off) red[tid] += red[tid + off]; __syncthreads(); }
    float mu = red[0] / DIM_; __syncthreads();
    float d = (tid < DIM_) ? (v - mu) : 0.0f;
    red[tid] = d * d; __syncthreads();
    for (int off = 128; off >= 1; off >>= 1) { if (tid < off) red[tid] += red[tid + off]; __syncthreads(); }
    float var = red[0] / DIM_;
    if (tid < DIM_) {
        float y = (v - mu) * rsqrtf(var + 1e-5f) * w[tid] + b[tid];
        outh[row * DIM_ + tid] = (_Float16)y;
    }
}

// ---------------------------------------------------------------- generic WMMA GEMM: C = A(f16,MxK) @ W(f16,KxN) + bias
__global__ __launch_bounds__(256)
void wmma_gemm_kernel(const _Float16* __restrict__ A, const _Float16* __restrict__ W,
                      const float* __restrict__ bias, float* __restrict__ C,
                      _Float16* __restrict__ Ch, int M, int K, int Ncols) {
    const int lane  = threadIdx.x & 31;
    const int wave  = threadIdx.x >> 5;
    const int mtile = blockIdx.x;
    const int ntile = blockIdx.y * 8 + wave;
    if (ntile >= (Ncols >> 4)) return;               // wave-uniform exit
    const int ml = lane & 15;
    const int kb = (lane >> 4) * 8;
    const int col = ntile * 16 + ml;
    const long rowA = (long)(mtile * 16 + ml) * K;
    v8f acc = {};
    for (int kk = 0; kk < K; kk += 32) {
        if (kk + 32 < K) __builtin_prefetch(&W[(long)(kk + 32) * Ncols + col], 0, 1);
        v16h a, bfr;
#pragma unroll
        for (int e = 0; e < 16; ++e) {
            int k = kk + kb + e + ((e >= 8) ? 8 : 0);
            a[e]   = A[rowA + k];
            bfr[e] = W[(long)k * Ncols + col];
        }
        acc = __builtin_amdgcn_wmma_f32_16x16x32_f16(false, a, false, bfr, (short)0, acc, false, false);
    }
    const float bv = bias ? bias[col] : 0.0f;
#pragma unroll
    for (int r = 0; r < 8; ++r) {
        long row = (long)mtile * 16 + r + 8 * (lane >> 4);
        float v = acc[r] + bv;
        C[row * Ncols + col] = v;
        if (Ch) Ch[row * Ncols + col] = (_Float16)v;
    }
}

// ---------------------------------------------------------------- RD=20 projection + L2 row-normalize (qa / ka)
__global__ void rdproj_kernel(const _Float16* __restrict__ A, const float* __restrict__ w,
                              const float* __restrict__ bias, float* __restrict__ out, int rows) {
    int row = blockIdx.x * blockDim.x + threadIdx.x;
    if (row >= rows) return;
    float acc[RD_];
#pragma unroll
    for (int r = 0; r < RD_; ++r) acc[r] = bias[r];
    for (int k = 0; k < DIM_; ++k) {
        float a = (float)A[(long)row * DIM_ + k];
#pragma unroll
        for (int r = 0; r < RD_; ++r) acc[r] += a * w[k * RD_ + r];
    }
    float s = 0.0f;
#pragma unroll
    for (int r = 0; r < RD_; ++r) s += acc[r] * acc[r];
    float inv = 1.0f / fmaxf(sqrtf(s), 1e-12f);
#pragma unroll
    for (int r = 0; r < RD_; ++r) out[(long)row * RD_ + r] = acc[r] * inv;
}

// ---------------------------------------------------------------- sim = softmax_t(qa.ka * (1+s_t*logNT)); also tk_id
__global__ __launch_bounds__(128)
void sim_kernel(const float* __restrict__ qa, const float* __restrict__ ka,
                const float* __restrict__ atd_scale, float* __restrict__ sim,
                _Float16* __restrict__ sim_h, int* __restrict__ tk_id) {
    __shared__ float sl[NT_];
    __shared__ float red[NT_];
    long rown = blockIdx.x;            // b*N + n
    int b = (int)(rown / N_);
    int t = threadIdx.x;
    float s = fminf(fmaxf(atd_scale[t], 0.0f), 1.0f);
    float dot = 0.0f;
#pragma unroll
    for (int r = 0; r < RD_; ++r) dot += qa[rown * RD_ + r] * ka[(b * NT_ + t) * RD_ + r];
    float logit = dot * (1.0f + s * 4.852030263919617f);    // log(128)
    sl[t] = logit; red[t] = logit; __syncthreads();
    for (int off = 64; off >= 1; off >>= 1) { if (t < off) red[t] = fmaxf(red[t], red[t + off]); __syncthreads(); }
    float mx = red[0]; __syncthreads();
    float e = __expf(logit - mx);
    red[t] = e; __syncthreads();
    for (int off = 64; off >= 1; off >>= 1) { if (t < off) red[t] += red[t + off]; __syncthreads(); }
    float p = e / red[0];
    sim[rown * NT_ + t]   = p;
    sim_h[rown * NT_ + t] = (_Float16)p;
    if (t == 0) {
        int am = 0; float bv = sl[0];
        for (int j = 1; j < NT_; ++j) if (sl[j] > bv) { bv = sl[j]; am = j; }
        tk_id[rown] = am;
    }
}

// ---------------------------------------------------------------- counting sort pieces
__global__ void hist_kernel(const int* __restrict__ tk, int* __restrict__ hist) {
    int i = blockIdx.x * blockDim.x + threadIdx.x;
    if (i >= BN_) return;
    atomicAdd(&hist[(i / N_) * NT_ + tk[i]], 1);
}
__global__ void scan_kernel(const int* __restrict__ hist, int* __restrict__ off) {
    int b = threadIdx.x;
    if (b < B_) {
        int acc = 0;
        for (int t = 0; t < NT_; ++t) { off[b * NT_ + t] = acc; acc += hist[b * NT_ + t]; }
    }
}
__global__ void pos_kernel(const int* __restrict__ tk, const int* __restrict__ off, int* __restrict__ pos) {
    int i = blockIdx.x * blockDim.x + threadIdx.x;
    if (i >= BN_) return;
    int b = i / N_, n = i % N_;
    int key = tk[i];
    const int* tkb = tk + (long)b * N_;
    int rank = 0;
    for (int j = 0; j < n; ++j) rank += (tkb[j] == key);   // stable rank
    pos[i] = off[b * NT_ + key] + rank;
}
__global__ void aca_gather_kernel(const float* __restrict__ qkv, const int* __restrict__ pos,
                                  _Float16* __restrict__ outh) {
    long idx = (long)blockIdx.x * blockDim.x + threadIdx.x;
    if (idx >= (long)BN_ * C3_) return;
    int  c   = (int)(idx % C3_);
    long row = idx / C3_;
    int  b   = (int)(row / N_);
    int  p   = pos[row];
    outh[((long)b * N_ + p) * C3_ + c] = (_Float16)qkv[idx];
}
__global__ void aca_scatter_kernel(const float* __restrict__ og, const int* __restrict__ pos,
                                   _Float16* __restrict__ outh) {
    long idx = (long)blockIdx.x * blockDim.x + threadIdx.x;
    if (idx >= (long)BN_ * DIM_) return;
    int  c   = (int)(idx % DIM_);
    long row = idx / DIM_;
    int  b   = (int)(row / N_);
    int  p   = pos[row];
    outh[idx] = (_Float16)og[((long)b * N_ + p) * DIM_ + c];
}

// ---------------------------------------------------------------- 256-token attention (window MSA + category MSA)
// one block = one (b, window/group, head); 8 waves, each wave owns two 16-row Q tiles.
// mode 0: load f32 qkv with shifted-window mapping (VALU convert).
// mode 1: sorted f16 qkv -> Q/K via global_load_async_to_lds_b128 (ASYNCcnt),
//         V via Tensor Data Mover tensor_load_to_lds (TENSORcnt).
__global__ __launch_bounds__(256)
void attn256_kernel(int mode, const float* __restrict__ qkv, const _Float16* __restrict__ qkv_sorted,
                    const float* __restrict__ rpb_table, const int* __restrict__ rpi,
                    const float* __restrict__ attn_mask, const float* __restrict__ logit_scale,
                    float* __restrict__ out) {
    __shared__ __attribute__((aligned(16))) _Float16 Qs[NW2_][HD_];
    __shared__ __attribute__((aligned(16))) _Float16 Ks[NW2_][HD_];
    __shared__ __attribute__((aligned(16))) _Float16 Vs[NW2_][HD_];
    __shared__ __attribute__((aligned(16))) _Float16 Ps[8][16][32];
    int bx   = blockIdx.x;
    int head = bx % HEADS_;
    int grp  = (bx / HEADS_) % NWIN_;
    int b    = bx / (HEADS_ * NWIN_);
    int tid  = threadIdx.x;
    float scale;
    if (mode == 0) {                               // shifted-window load (roll by -SHIFT), f32 -> f16
        int wy = grp >> 3, wx = grp & 7;
        for (int idx = tid; idx < NW2_ * HD_; idx += 256) {
            int tok = idx >> 5, d = idx & 31;
            int y  = (wy * WS_ + (tok >> 4) + SHIFT_) & (H_ - 1);
            int xx = (wx * WS_ + (tok & 15) + SHIFT_) & (W_ - 1);
            long base = ((long)b * N_ + y * W_ + xx) * C3_ + head * HD_ + d;
            Qs[tok][d] = (_Float16)qkv[base];
            Ks[tok][d] = (_Float16)qkv[base + DIM_];
            Vs[tok][d] = (_Float16)qkv[base + 2 * DIM_];
        }
        scale = 0.17677669529663687f;              // 1/sqrt(32)
    } else {                                       // category groups, f16 source: async + TDM copies
        unsigned ldsQ = (unsigned)(uintptr_t)&Qs[0][0];
        unsigned ldsK = (unsigned)(uintptr_t)&Ks[0][0];
        unsigned ldsV = (unsigned)(uintptr_t)&Vs[0][0];
        long rowbase0 = (long)b * N_ + grp * NW2_;
        // Q and K: per-lane async 16-byte chunks  (256 tokens x 64B each)
        for (int c = tid; c < NW2_ * 4; c += 256) {
            int tok = c >> 2, q = c & 3;
            const _Float16* gq = qkv_sorted + (rowbase0 + tok) * C3_ + head * HD_ + q * 8;
            unsigned long long aq = (unsigned long long)(uintptr_t)gq;
            unsigned long long ak = aq + (unsigned long long)(DIM_ * 2);
            unsigned lq = ldsQ + tok * 64 + q * 16;
            unsigned lk = ldsK + tok * 64 + q * 16;
            asm volatile("global_load_async_to_lds_b128 %0, %1, off" :: "v"(lq), "v"(aq) : "memory");
            asm volatile("global_load_async_to_lds_b128 %0, %1, off" :: "v"(lk), "v"(ak) : "memory");
        }
        asm volatile("s_wait_asynccnt 0" ::: "memory");
        // V: one 2-D TDM descriptor, issued by wave 0 only (tile 32x256 halfs, row stride 576 halfs)
        if (tid < 32) {
            const _Float16* gv = qkv_sorted + rowbase0 * C3_ + 2 * DIM_ + head * HD_;
            unsigned long long ga = (unsigned long long)(uintptr_t)gv;
            int w0 = __builtin_amdgcn_readfirstlane(1);                           // count=1
            int w1 = __builtin_amdgcn_readfirstlane((int)ldsV);                   // lds_addr
            int w2 = __builtin_amdgcn_readfirstlane((int)(ga & 0xffffffffu));     // global_addr lo
            int w3 = __builtin_amdgcn_readfirstlane((int)(((ga >> 32) & 0x1ffffffull) | 0x80000000u)); // hi | type=2
            v4i g0 = { w0, w1, w2, w3 };
            int u0 = __builtin_amdgcn_readfirstlane(0x10000);       // data_size = 2B
            int u1 = __builtin_amdgcn_readfirstlane(32 << 16);      // tensor_dim0 = 32 (low16 in [31:16])
            int u2 = __builtin_amdgcn_readfirstlane(256 << 16);     // tensor_dim1 = 256 (low16 in [31:16])
            int u3 = __builtin_amdgcn_readfirstlane(32 << 16);      // tile_dim0 = 32
            int u4 = __builtin_amdgcn_readfirstlane(256);           // tile_dim1 = 256, tile_dim2 = 0
            int u5 = __builtin_amdgcn_readfirstlane(576);           // tensor_dim0_stride = 576
            int u6 = __builtin_amdgcn_readfirstlane(0);
            int u7 = __builtin_amdgcn_readfirstlane(0);
            v8i g1 = { u0, u1, u2, u3, u4, u5, u6, u7 };
            asm volatile("tensor_load_to_lds %0, %1" :: "s"(g0), "s"(g1) : "memory");
#if __has_builtin(__builtin_amdgcn_s_wait_tensorcnt)
            __builtin_amdgcn_s_wait_tensorcnt(0);
#else
            asm volatile("s_wait_tensorcnt 0" ::: "memory");
#endif
        }
        scale = __expf(fminf(logit_scale[0], 4.6051702f));   // exp(min(ls, log 100))
    }
    __syncthreads();

    int lane = tid & 31, wave = tid >> 5;
    int ml = lane & 15, kb = (lane >> 4) * 8, hi = lane >> 4;
    for (int qt = wave * 2; qt < wave * 2 + 2; ++qt) {
        v16h aQ = load_frag_lds(&Qs[qt * 16 + ml][0], kb);
        v8f sc[16];
        for (int jt = 0; jt < 16; ++jt) {
            v16h bK = load_frag_lds(&Ks[jt * 16 + ml][0], kb);
            v8f z = {};
            sc[jt] = __builtin_amdgcn_wmma_f32_16x16x32_f16(false, aQ, false, bK, (short)0, z, false, false);
#pragma unroll
            for (int r = 0; r < 8; ++r) {
                float v = sc[jt][r] * scale;
                if (mode == 0) {
                    int i = qt * 16 + r + 8 * hi;
                    int j = jt * 16 + ml;
                    v += rpb_table[rpi[i * NW2_ + j] * HEADS_ + head]
                       + attn_mask[(long)grp * NW2_ * NW2_ + i * NW2_ + j];
                }
                sc[jt][r] = v;
            }
        }
        // row softmax: rows live across lanes 0-15 (m=r) and 16-31 (m=r+8)
        float mx[8], inv[8];
#pragma unroll
        for (int r = 0; r < 8; ++r) {
            float v = -3.0e38f;
            for (int jt = 0; jt < 16; ++jt) v = fmaxf(v, sc[jt][r]);
            for (int off = 8; off >= 1; off >>= 1) v = fmaxf(v, __shfl_xor(v, off, 16));
            mx[r] = v;
        }
#pragma unroll
        for (int r = 0; r < 8; ++r) {
            float s = 0.0f;
            for (int jt = 0; jt < 16; ++jt) { float e = __expf(sc[jt][r] - mx[r]); sc[jt][r] = e; s += e; }
            for (int off = 8; off >= 1; off >>= 1) s += __shfl_xor(s, off, 16);
            inv[r] = 1.0f / s;
        }
        // O = P @ V : relayout P (C-layout f32 -> A-layout f16) through per-wave LDS staging
        v8f accO0 = {}, accO1 = {};
        for (int jp = 0; jp < 8; ++jp) {
#pragma unroll
            for (int half = 0; half < 2; ++half) {
                int jt = jp * 2 + half;
#pragma unroll
                for (int r = 0; r < 8; ++r)
                    Ps[wave][r + 8 * hi][half * 16 + ml] = (_Float16)(sc[jt][r] * inv[r]);
            }
            v16h aP = load_frag_lds(&Ps[wave][ml][0], kb);
            v16h bV0, bV1;
#pragma unroll
            for (int e = 0; e < 16; ++e) {
                int k = kb + e + ((e >= 8) ? 8 : 0);
                bV0[e] = Vs[jp * 32 + k][ml];
                bV1[e] = Vs[jp * 32 + k][16 + ml];
            }
            accO0 = __builtin_amdgcn_wmma_f32_16x16x32_f16(false, aP, false, bV0, (short)0, accO0, false, false);
            accO1 = __builtin_amdgcn_wmma_f32_16x16x32_f16(false, aP, false, bV1, (short)0, accO1, false, false);
        }
        long rowbase = (long)b * N_ + grp * NW2_;
#pragma unroll
        for (int r = 0; r < 8; ++r) {
            long row = rowbase + qt * 16 + r + 8 * hi;
            out[row * DIM_ + head * HD_ + ml]      = accO0[r];
            out[row * DIM_ + head * HD_ + 16 + ml] = accO1[r];
        }
    }
}

// ---------------------------------------------------------------- residual combine (with window inverse mapping)
__global__ void combine_kernel(const float* __restrict__ x, const float* __restrict__ winproj,
                               const float* __restrict__ x_atd, const float* __restrict__ x_aca,
                               float* __restrict__ x_mid) {
    long idx = (long)blockIdx.x * blockDim.x + threadIdx.x;
    if (idx >= (long)BN_ * DIM_) return;
    int  c   = (int)(idx % DIM_);
    long row = idx / DIM_;
    int  n   = (int)(row % N_);
    int  b   = (int)(row / N_);
    int  y = n / W_, xx = n % W_;
    int  yp = (y - SHIFT_) & (H_ - 1), xp = (xx - SHIFT_) & (W_ - 1);
    int  wy = yp >> 4, iy = yp & 15, wx = xp >> 4, ix = xp & 15;
    long wrow = ((long)b * NWIN_ + wy * 8 + wx) * NW2_ + iy * 16 + ix;
    x_mid[idx] = x[idx] + winproj[wrow * DIM_ + c] + x_atd[idx] + x_aca[idx];
}

// ---------------------------------------------------------------- elementwise gelu (in place)
__global__ void gelu_kernel(float* __restrict__ a, long n) {
    long i = (long)blockIdx.x * blockDim.x + threadIdx.x;
    if (i < n) a[i] = gelu_exact(a[i]);
}

// ---------------------------------------------------------------- depthwise 5x5 conv + gelu, output f16 (b,n,HID)
__global__ void dwconv_kernel(const float* __restrict__ hdn, const float* __restrict__ dw_w,
                              const float* __restrict__ dw_b, _Float16* __restrict__ outh) {
    long idx = (long)blockIdx.x * blockDim.x + threadIdx.x;
    if (idx >= (long)BN_ * HID_) return;
    int  c   = (int)(idx % HID_);
    long pos = idx / HID_;
    int  xx  = (int)(pos % W_);
    int  y   = (int)((pos / W_) % H_);
    int  b   = (int)(pos / N_);
    float acc = dw_b[c];
#pragma unroll
    for (int ky = 0; ky < 5; ++ky) {
        int yy = y + ky - 2;
        if (yy < 0 || yy >= H_) continue;
#pragma unroll
        for (int kx = 0; kx < 5; ++kx) {
            int xc = xx + kx - 2;
            if (xc < 0 || xc >= W_) continue;
            acc += hdn[((long)b * N_ + yy * W_ + xc) * HID_ + c] * dw_w[c * 25 + ky * 5 + kx];
        }
    }
    outh[idx] = (_Float16)gelu_exact(acc);
}

__global__ void add_kernel(const float* __restrict__ a, const float* __restrict__ b,
                           float* __restrict__ out, long n) {
    long i = (long)blockIdx.x * blockDim.x + threadIdx.x;
    if (i < n) out[i] = a[i] + b[i];
}

// ---------------------------------------------------------------- token dictionary refinement + instance norm
__global__ __launch_bounds__(256)
void td_refine_kernel(const float* __restrict__ sim, const float* __restrict__ xfin,
                      const float* __restrict__ td, const float* __restrict__ sigma,
                      const float* __restrict__ in3_w, const float* __restrict__ in3_b,
                      float* __restrict__ out_td) {
    __shared__ float red[256];
    int bt = blockIdx.x;               // b*NT + t
    int b = bt / NT_, t = bt % NT_;
    int tid = threadIdx.x;
    float mx = -3.0e38f;
    for (int n = tid; n < N_; n += 256) mx = fmaxf(mx, sim[((long)b * N_ + n) * NT_ + t]);
    red[tid] = mx; __syncthreads();
    for (int off = 128; off >= 1; off >>= 1) { if (tid < off) red[tid] = fmaxf(red[tid], red[tid + off]); __syncthreads(); }
    mx = red[0]; __syncthreads();
    float s = 0.0f;
    for (int n = tid; n < N_; n += 256) s += __expf(sim[((long)b * N_ + n) * NT_ + t] - mx);
    red[tid] = s; __syncthreads();
    for (int off = 128; off >= 1; off >>= 1) { if (tid < off) red[tid] += red[tid + off]; __syncthreads(); }
    float inv = 1.0f / red[0]; __syncthreads();
    float acc = 0.0f;
    if (tid < DIM_) {
        for (int n = 0; n < N_; ++n) {
            float w = __expf(sim[((long)b * N_ + n) * NT_ + t] - mx) * inv;
            acc += w * xfin[((long)b * N_ + n) * DIM_ + tid];
        }
        float sg = sigma[t];
        acc = sg * td[((long)b * NT_ + t) * DIM_ + tid] + (1.0f - sg) * acc;
    }
    red[tid] = (tid < DIM_) ? acc : 0.0f; __syncthreads();
    for (int off = 128; off >= 1; off >>= 1) { if (tid < off) red[tid] += red[tid + off]; __syncthreads(); }
    float mu = red[0] / DIM_; __syncthreads();
    float d = (tid < DIM_) ? (acc - mu) : 0.0f;
    red[tid] = d * d; __syncthreads();
    for (int off = 128; off >= 1; off >>= 1) { if (tid < off) red[tid] += red[tid + off]; __syncthreads(); }
    float var = red[0] / DIM_;
    if (tid < DIM_)
        out_td[((long)b * NT_ + t) * DIM_ + tid] = (acc - mu) * rsqrtf(var + 1e-5f) * in3_w[t] + in3_b[t];
}

// ================================================================ host
static void launch_gemm(const _Float16* A, const _Float16* W, const float* bias,
                        float* C, _Float16* Ch, int M, int K, int Ncols, hipStream_t s) {
    dim3 g(M / 16, (Ncols / 16 + 7) / 8);
    wmma_gemm_kernel<<<g, dim3(256), 0, s>>>(A, W, bias, C, Ch, M, K, Ncols);
}
static void launch_cvt(const float* src, _Float16* dst, long n, hipStream_t s) {
    cvt_f32_f16_kernel<<<(int)((n + 255) / 256), 256, 0, s>>>(src, dst, n);
}

extern "C" void kernel_launch(void* const* d_in, const int* in_sizes, int n_in,
                              void* d_out, int out_size, void* d_ws, size_t ws_size,
                              hipStream_t stream) {
    const float* x          = (const float*)d_in[0];
    const float* td         = (const float*)d_in[1];
    const float* attn_mask  = (const float*)d_in[2];
    const float* wqkv_w     = (const float*)d_in[3];
    const float* wqkv_b     = (const float*)d_in[4];
    const float* rpb_table  = (const float*)d_in[5];
    const float* win_proj_w = (const float*)d_in[6];
    const float* win_proj_b = (const float*)d_in[7];
    const float* wq_w       = (const float*)d_in[8];
    const float* wq_b       = (const float*)d_in[9];
    const float* wk_w       = (const float*)d_in[10];
    const float* wk_b       = (const float*)d_in[11];
    const float* wv_w       = (const float*)d_in[12];
    const float* wv_b       = (const float*)d_in[13];
    const float* atd_scale  = (const float*)d_in[14];
    const float* aca_proj_w = (const float*)d_in[15];
    const float* aca_proj_b = (const float*)d_in[16];
    const float* logit_scale= (const float*)d_in[17];
    const float* fc1_w      = (const float*)d_in[18];
    const float* fc1_b      = (const float*)d_in[19];
    const float* dw_w       = (const float*)d_in[20];
    const float* dw_b       = (const float*)d_in[21];
    const float* fc2_w      = (const float*)d_in[22];
    const float* fc2_b      = (const float*)d_in[23];
    const float* ln1_w      = (const float*)d_in[24];
    const float* ln1_b      = (const float*)d_in[25];
    const float* ln2_w      = (const float*)d_in[26];
    const float* ln2_b      = (const float*)d_in[27];
    const float* in3_w      = (const float*)d_in[28];
    const float* in3_b      = (const float*)d_in[29];
    const float* sigma      = (const float*)d_in[30];
    const int*   rpi        = (const int*)d_in[31];

    float* x_out  = (float*)d_out;                       // (B,N,DIM)
    float* td_out = x_out + (long)BN_ * DIM_;            // (B,NT,DIM)

    // ---- workspace bump allocator ----
    size_t cur = 0;
    auto alloc = [&](size_t bytes) -> void* {
        void* p = (char*)d_ws + cur;
        cur += (bytes + 255) & ~(size_t)255;
        return p;
    };
    _Float16* xn_h     = (_Float16*)alloc((size_t)BN_ * DIM_ * 2);
    float*    qkv      = (float*)   alloc((size_t)BN_ * C3_ * 4);
    _Float16* wqkv_h   = (_Float16*)alloc((size_t)DIM_ * C3_ * 2);
    _Float16* winp_h   = (_Float16*)alloc((size_t)DIM_ * DIM_ * 2);
    _Float16* wv_h     = (_Float16*)alloc((size_t)DIM_ * DIM_ * 2);
    _Float16* acap_h   = (_Float16*)alloc((size_t)DIM_ * DIM_ * 2);
    _Float16* fc1_h    = (_Float16*)alloc((size_t)DIM_ * HID_ * 2);
    _Float16* fc2_h    = (_Float16*)alloc((size_t)HID_ * DIM_ * 2);
    _Float16* td_h     = (_Float16*)alloc((size_t)B_ * NT_ * DIM_ * 2);
    float*    qa       = (float*)   alloc((size_t)BN_ * RD_ * 4);
    float*    ka       = (float*)   alloc((size_t)B_ * NT_ * RD_ * 4);
    float*    va       = (float*)   alloc((size_t)B_ * NT_ * DIM_ * 4);
    _Float16* va_h     = (_Float16*)alloc((size_t)B_ * NT_ * DIM_ * 2);
    float*    sim      = (float*)   alloc((size_t)BN_ * NT_ * 4);
    _Float16* sim_h    = (_Float16*)alloc((size_t)BN_ * NT_ * 2);
    int*      tk_id    = (int*)     alloc((size_t)BN_ * 4);
    int*      hist     = (int*)     alloc((size_t)B_ * NT_ * 4);
    int*      offs     = (int*)     alloc((size_t)B_ * NT_ * 4);
    int*      pos      = (int*)     alloc((size_t)BN_ * 4);
    _Float16* qkv_srt  = (_Float16*)alloc((size_t)BN_ * C3_ * 2);
    float*    o_win    = (float*)   alloc((size_t)BN_ * DIM_ * 4);
    _Float16* o_win_h  = (_Float16*)alloc((size_t)BN_ * DIM_ * 2);
    float*    winproj  = (float*)   alloc((size_t)BN_ * DIM_ * 4);
    float*    og       = (float*)   alloc((size_t)BN_ * DIM_ * 4);
    _Float16* og_h     = (_Float16*)alloc((size_t)BN_ * DIM_ * 2);
    float*    x_aca    = (float*)   alloc((size_t)BN_ * DIM_ * 4);
    float*    x_atd    = (float*)   alloc((size_t)BN_ * DIM_ * 4);
    float*    x_mid    = (float*)   alloc((size_t)BN_ * DIM_ * 4);
    _Float16* xn2_h    = (_Float16*)alloc((size_t)BN_ * DIM_ * 2);
    float*    fc1out   = (float*)   alloc((size_t)BN_ * HID_ * 4);
    _Float16* conv_h   = (_Float16*)alloc((size_t)BN_ * HID_ * 2);
    float*    fc2out   = (float*)   alloc((size_t)BN_ * DIM_ * 4);

    // ---- weight / dict conversions to f16 ----
    launch_cvt(wqkv_w,     wqkv_h, (long)DIM_ * C3_,  stream);
    launch_cvt(win_proj_w, winp_h, (long)DIM_ * DIM_, stream);
    launch_cvt(wv_w,       wv_h,   (long)DIM_ * DIM_, stream);
    launch_cvt(aca_proj_w, acap_h, (long)DIM_ * DIM_, stream);
    launch_cvt(fc1_w,      fc1_h,  (long)DIM_ * HID_, stream);
    launch_cvt(fc2_w,      fc2_h,  (long)HID_ * DIM_, stream);
    launch_cvt(td,         td_h,   (long)B_ * NT_ * DIM_, stream);

    // ---- LN1 + QKV projection (WMMA) ----
    ln_kernel<<<BN_, 256, 0, stream>>>(x, ln1_w, ln1_b, xn_h);
    launch_gemm(xn_h, wqkv_h, wqkv_b, qkv, nullptr, BN_, DIM_, C3_, stream);

    // ---- dictionary cross-attention (ATD) ----
    rdproj_kernel<<<(BN_ + 255) / 256, 256, 0, stream>>>(xn_h, wq_w, wq_b, qa, BN_);
    rdproj_kernel<<<1, 256, 0, stream>>>(td_h, wk_w, wk_b, ka, B_ * NT_);
    launch_gemm(td_h, wv_h, wv_b, va, va_h, B_ * NT_, DIM_, DIM_, stream);
    sim_kernel<<<BN_, 128, 0, stream>>>(qa, ka, atd_scale, sim, sim_h, tk_id);
    for (int b = 0; b < B_; ++b)
        launch_gemm(sim_h + (long)b * N_ * NT_, va_h + (long)b * NT_ * DIM_, nullptr,
                    x_atd + (long)b * N_ * DIM_, nullptr, N_, NT_, DIM_, stream);

    // ---- category sort (counting sort, stable) + gather ----
    hipMemsetAsync(hist, 0, (size_t)B_ * NT_ * 4, stream);
    hist_kernel<<<(BN_ + 255) / 256, 256, 0, stream>>>(tk_id, hist);
    scan_kernel<<<1, 64, 0, stream>>>(hist, offs);
    pos_kernel<<<(BN_ + 255) / 256, 256, 0, stream>>>(tk_id, offs, pos);
    aca_gather_kernel<<<(int)(((long)BN_ * C3_ + 255) / 256), 256, 0, stream>>>(qkv, pos, qkv_srt);

    // ---- window MSA + category MSA (WMMA attention) ----
    attn256_kernel<<<B_ * NWIN_ * HEADS_, 256, 0, stream>>>(
        0, qkv, qkv_srt, rpb_table, rpi, attn_mask, logit_scale, o_win);
    launch_cvt(o_win, o_win_h, (long)BN_ * DIM_, stream);
    launch_gemm(o_win_h, winp_h, win_proj_b, winproj, nullptr, BN_, DIM_, DIM_, stream);

    attn256_kernel<<<B_ * NWIN_ * HEADS_, 256, 0, stream>>>(
        1, qkv, qkv_srt, rpb_table, rpi, attn_mask, logit_scale, og);
    aca_scatter_kernel<<<(int)(((long)BN_ * DIM_ + 255) / 256), 256, 0, stream>>>(og, pos, og_h);
    launch_gemm(og_h, acap_h, aca_proj_b, x_aca, nullptr, BN_, DIM_, DIM_, stream);

    // ---- residual combine ----
    combine_kernel<<<(int)(((long)BN_ * DIM_ + 255) / 256), 256, 0, stream>>>(x, winproj, x_atd, x_aca, x_mid);

    // ---- ConvFFN ----
    ln_kernel<<<BN_, 256, 0, stream>>>(x_mid, ln2_w, ln2_b, xn2_h);
    launch_gemm(xn2_h, fc1_h, fc1_b, fc1out, nullptr, BN_, DIM_, HID_, stream);
    gelu_kernel<<<(int)(((long)BN_ * HID_ + 255) / 256), 256, 0, stream>>>(fc1out, (long)BN_ * HID_);
    dwconv_kernel<<<(int)(((long)BN_ * HID_ + 255) / 256), 256, 0, stream>>>(fc1out, dw_w, dw_b, conv_h);
    launch_gemm(conv_h, fc2_h, fc2_b, fc2out, nullptr, BN_, HID_, DIM_, stream);
    add_kernel<<<(int)(((long)BN_ * DIM_ + 255) / 256), 256, 0, stream>>>(x_mid, fc2out, x_out, (long)BN_ * DIM_);

    // ---- token dictionary refinement ----
    td_refine_kernel<<<B_ * NT_, 256, 0, stream>>>(sim, x_out, td, sigma, in3_w, in3_b, td_out);

    (void)in_sizes; (void)n_in; (void)out_size; (void)ws_size;
}
